// FullContrastiveLoss_61795989455419
// MI455X (gfx1250) — matrix operand order repaired and checked by reading, not compiled, not run on hardware
//
#include <hip/hip_runtime.h>

#define BATCH 8192
#define DIM   1024
#define TEMP_INV 2.0f   // 1 / 0.5
#define EPS_F 1e-8f
#define TILE  256
#define NT_TILES (BATCH / TILE)                    // 32
#define NBLK (NT_TILES * (NT_TILES + 1) / 2)       // 528 triangle blocks

typedef __attribute__((ext_vector_type(16))) __bf16 v16bf;
typedef __attribute__((ext_vector_type(8)))  float  v8f;

union FragU { uint4 u[2]; v16bf v; };

__device__ __forceinline__ unsigned short f32_to_bf16(float f) {
  unsigned int u = __float_as_uint(f);
  u += 0x7FFFu + ((u >> 16) & 1u);   // round-to-nearest-even (inputs finite)
  return (unsigned short)(u >> 16);
}

// one async memory->LDS 16B copy (ASYNCcnt-tracked, bypasses VGPRs)
__device__ __forceinline__ void async_ld_b128(unsigned ldsOff, const void* gaddr) {
  asm volatile("global_load_async_to_lds_b128 %0, %1, off"
               :: "v"(ldsOff), "v"(gaddr) : "memory");
}
__device__ __forceinline__ void wait_async0() {
  asm volatile("s_wait_asynccnt 0x0" ::: "memory");
}

// ---------------------------------------------------------------------------
// Kernel 1: L2-normalize rows (f32) -> bf16 feature matrix; zero accumulators.
// grid = 2*8192, block = 256.
// ---------------------------------------------------------------------------
__global__ void fcl_normalize(const float* __restrict__ text,
                              const float* __restrict__ image,
                              unsigned short* __restrict__ Fn,
                              float* __restrict__ rowSame,
                              float* __restrict__ rowAll) {
  const int b   = blockIdx.x;            // feature*8192 + row
  const int tid = threadIdx.x;           // each handles 4 floats
  const float* src = ((b >> 13) ? image : text) + (size_t)(b & (BATCH - 1)) * DIM;

  float4 v = ((const float4*)src)[tid];
  float s = v.x * v.x + v.y * v.y + v.z * v.z + v.w * v.w;
#pragma unroll
  for (int off = 16; off >= 1; off >>= 1) s += __shfl_xor(s, off, 32);

  __shared__ float wsum[8];
  __shared__ float sInv;
  if ((tid & 31) == 0) wsum[tid >> 5] = s;
  __syncthreads();
  if (tid == 0) {
    float t = 0.f;
#pragma unroll
    for (int i = 0; i < 8; ++i) t += wsum[i];
    sInv = 1.0f / fmaxf(sqrtf(t), 1e-12f);
    rowSame[b] = 0.f;
    rowAll[b]  = 0.f;
  }
  __syncthreads();
  const float iv = sInv;

  unsigned int lo = (unsigned int)f32_to_bf16(v.x * iv) |
                    ((unsigned int)f32_to_bf16(v.y * iv) << 16);
  unsigned int hi = (unsigned int)f32_to_bf16(v.z * iv) |
                    ((unsigned int)f32_to_bf16(v.w * iv) << 16);
  *(uint2*)(Fn + (size_t)b * DIM + tid * 4) = make_uint2(lo, hi);
}

// ---------------------------------------------------------------------------
// Kernel 2: symmetric fused Gram + exp + masked row/col sums via WMMA bf16.
// Only upper-triangle 256x256 tile blocks are computed; off-diagonal blocks
// scatter exp-sums to rows AND (by symmetry) to columns -> ~2x less GEMM work.
// 512 threads = 16 waves (4x4); wave tile 64x64 = 4x4 WMMA accumulators.
// LDS double-buffer filled with GLOBAL_LOAD_ASYNC_TO_LDS_B128 (ASYNCcnt).
// grid = (528, 2), block = 512.
// ---------------------------------------------------------------------------
__global__ void __launch_bounds__(512)
fcl_gram_wmma(const unsigned short* __restrict__ Fn,
              const int* __restrict__ label,
              float* __restrict__ rowSame,
              float* __restrict__ rowAll) {
  const int feat = blockIdx.y;
  const unsigned short* F = Fn + (size_t)feat * BATCH * DIM;
  float* rs = rowSame + feat * BATCH;
  float* ra = rowAll  + feat * BATCH;

  // triangle block index -> (ti, tj) with ti <= tj
  int ti = 0, rem = blockIdx.x;
  while (rem >= NT_TILES - ti) { rem -= NT_TILES - ti; ++ti; }
  const int tj = ti + rem;
  const int Mbase = ti * TILE;
  const int Nbase = tj * TILE;
  const bool offdiag = (ti != tj);

  const int tid    = threadIdx.x;       // 0..511
  const int wave   = tid >> 5;          // 0..15
  const int lane   = tid & 31;
  const int half   = lane >> 4;
  const int lr     = lane & 15;
  const int wave_m = wave & 3;          // 4 waves along M
  const int wave_n = wave >> 2;         // 4 waves along N

  // double-buffered K-panels: A/B 256x32 bf16 per buffer (64 KB total)
  __shared__ __attribute__((aligned(16))) unsigned short shA[2][TILE][32];
  __shared__ __attribute__((aligned(16))) unsigned short shB[2][TILE][32];
  __shared__ unsigned char sRowL[TILE];
  __shared__ unsigned char sColL[TILE];

  if (tid < TILE) sRowL[tid]        = (unsigned char)(label[Mbase + tid] != 0);
  else            sColL[tid - TILE] = (unsigned char)(label[Nbase + (tid - TILE)] != 0);

  // cooperative copy: thread t moves 32B of A row (t>>1) and 32B of B row (t>>1)
  const int crow = tid >> 1;
  const int csub = (tid & 1) * 16;      // element offset (16 bf16 = 32B)
  const unsigned short* gA = F + (size_t)(Mbase + crow) * DIM + csub;
  const unsigned short* gB = F + (size_t)(Nbase + crow) * DIM + csub;
  const unsigned ldsA0 = (unsigned)(unsigned long long)&shA[0][crow][csub];
  const unsigned ldsB0 = (unsigned)(unsigned long long)&shB[0][crow][csub];
  const unsigned strA = (unsigned)sizeof(shA[0]);

  auto issue_copy = [&](int buf, int k) {
    const unsigned stride = (unsigned)buf * strA;
    const unsigned short* pa = gA + k;
    async_ld_b128(ldsA0 + stride,       pa);
    async_ld_b128(ldsA0 + stride + 16u, pa + 8);
    const unsigned short* pb = gB + k;
    async_ld_b128(ldsB0 + stride,       pb);
    async_ld_b128(ldsB0 + stride + 16u, pb + 8);
  };

  v8f c[4][4] = {};

  issue_copy(0, 0);

  for (int ki = 0; ki < DIM / 32; ++ki) {
    const int buf = ki & 1;
    wait_async0();        // this wave's async copies for `buf` have landed
    __syncthreads();      // everyone's copies visible; prev reads of buf^1 done
    if (ki + 1 < DIM / 32) issue_copy(buf ^ 1, (ki + 1) * 32);

    v16bf a[4], b[4];
    // A 16x32 bf16: half 0 -> K{0..7,16..23}, half 1 -> K{8..15,24..31}
#pragma unroll
    for (int mt = 0; mt < 4; ++mt) {
      const uint4* p = (const uint4*)&shA[buf][wave_m * 64 + mt * 16 + lr][0];
      FragU t; t.u[0] = p[half]; t.u[1] = p[2 + half];
      a[mt] = t.v;
    }
    // B 32x16 bf16: lane = column; half 0 -> K{0..7,16..23}, half 1 -> K{8..15,24..31}
#pragma unroll
    for (int nt = 0; nt < 4; ++nt) {
      const uint4* p = (const uint4*)&shB[buf][wave_n * 64 + nt * 16 + lr][0];
      FragU t; t.u[0] = p[half]; t.u[1] = p[2 + half];
      b[nt] = t.v;
    }
#pragma unroll
    for (int mt = 0; mt < 4; ++mt)
#pragma unroll
      for (int nt = 0; nt < 4; ++nt)
        c[mt][nt] = __builtin_amdgcn_wmma_f32_16x16x32_bf16(
            false, a[mt], false, b[nt], (short)0, c[mt][nt], false, false);
  }

  // Epilogue: exp, diagonal/label masks, per-row sums; per-col sums if offdiag.
  // C layout: VGPR v, lanes 0-15 -> row v, lanes 16-31 -> row v+8; col = lr.
  const int col0 = wave_n * 64 + lr;
  float colAll[4]  = {0.f, 0.f, 0.f, 0.f};
  float colSame[4] = {0.f, 0.f, 0.f, 0.f};
#pragma unroll
  for (int mt = 0; mt < 4; ++mt) {
#pragma unroll
    for (int v = 0; v < 8; ++v) {
      const int row_local = wave_m * 64 + mt * 16 + v + 8 * half;
      const int row = Mbase + row_local;
      const unsigned char rl = sRowL[row_local];
      float sAll = 0.f, sSame = 0.f;
#pragma unroll
      for (int nt = 0; nt < 4; ++nt) {
        const int col_local = col0 + nt * 16;
        float g = c[mt][nt][v];
        float e = __expf(g * TEMP_INV);        // |g|<=~1 -> no overflow
        if (row == Nbase + col_local) e = 0.f; // exclude diagonal
        float es = (rl == sColL[col_local]) ? e : 0.f;
        sAll += e;  sSame += es;
        colAll[nt] += e;  colSame[nt] += es;
      }
#pragma unroll
      for (int off = 1; off < 16; off <<= 1) {
        sAll  += __shfl_xor(sAll,  off, 32);
        sSame += __shfl_xor(sSame, off, 32);
      }
      if (lr == 0) {
        atomicAdd(&ra[row], sAll);
        atomicAdd(&rs[row], sSame);
      }
    }
  }
  if (offdiag) {
    // transpose contribution: column j receives exp(G[i,j]) for all tile rows i
#pragma unroll
    for (int nt = 0; nt < 4; ++nt) {
      float ca = colAll[nt]  + __shfl_xor(colAll[nt],  16, 32);
      float cs = colSame[nt] + __shfl_xor(colSame[nt], 16, 32);
      if (half == 0) {
        const int col = Nbase + col0 + nt * 16;
        atomicAdd(&ra[col], ca);
        atomicAdd(&rs[col], cs);
      }
    }
  }
}

// ---------------------------------------------------------------------------
// Kernel 3: final scalar loss. grid = 1, block = 256.
// ---------------------------------------------------------------------------
__global__ void fcl_finalize(const float* __restrict__ rowSame,
                             const float* __restrict__ rowAll,
                             float* __restrict__ out) {
  float s = 0.f;
  for (int i = threadIdx.x; i < 2 * BATCH; i += 256) {
    float r = rowSame[i] / rowAll[i] + EPS_F;
    s += -logf(r);
  }
#pragma unroll
  for (int off = 16; off >= 1; off >>= 1) s += __shfl_xor(s, off, 32);
  __shared__ float wsum[8];
  if ((threadIdx.x & 31) == 0) wsum[threadIdx.x >> 5] = s;
  __syncthreads();
  if (threadIdx.x == 0) {
    float t = 0.f;
#pragma unroll
    for (int i = 0; i < 8; ++i) t += wsum[i];
    out[0] = t * (1.0f / 4096.0f);   // NUM_R == NUM_NR == 4096
  }
}

// ---------------------------------------------------------------------------
extern "C" void kernel_launch(void* const* d_in, const int* in_sizes, int n_in,
                              void* d_out, int out_size, void* d_ws, size_t ws_size,
                              hipStream_t stream) {
  const float* text  = (const float*)d_in[0];
  const float* image = (const float*)d_in[1];
  const int*   label = (const int*)d_in[2];

  // workspace: [2*8192*1024 bf16 features][rowSame f32 x16384][rowAll f32 x16384]
  unsigned short* Fn = (unsigned short*)d_ws;
  const size_t fnBytes = (size_t)2 * BATCH * DIM * sizeof(unsigned short);
  float* rowSame = (float*)((char*)d_ws + fnBytes);
  float* rowAll  = rowSame + 2 * BATCH;

  fcl_normalize<<<2 * BATCH, 256, 0, stream>>>(text, image, Fn, rowSame, rowAll);
  fcl_gram_wmma<<<dim3(NBLK, 2), 512, 0, stream>>>(Fn, label, rowSame, rowAll);
  fcl_finalize<<<1, 256, 0, stream>>>(rowSame, rowAll, (float*)d_out);
}
